// MoE_82910048682500
// MI455X (gfx1250) — matrix operand order repaired and checked by reading, not compiled, not run on hardware
//
#include <hip/hip_runtime.h>
#include <math.h>

typedef __attribute__((ext_vector_type(16))) __bf16 v16bf;
typedef __attribute__((ext_vector_type(8)))  float  v8f;

#define T_TOK 8192
#define MDIM  2048
#define FDIM  8192
#define NEXP  8
#define CAP   1024
#define LDA   40     // ushort stride for 32-deep K tiles (+8 pad, keeps 16B alignment)
#define BM    128
#define BN    256
#define BK    32

// native bf16 convert (v_cvt_*_bf16_f32, RTE) instead of manual integer rounding
__device__ __forceinline__ unsigned short f2bf(float f) {
  union { __bf16 h; unsigned short u; } v;
  v.h = (__bf16)f;
  return v.u;
}

// branch-free tanh-GELU: tanh(u) = 1 - 2/(1 + exp2(2u*log2e)); saturates correctly
__device__ __forceinline__ float gelu_fast(float x) {
  float u  = 0.7978845608028654f * (x + 0.044715f * x * x * x);
  float e  = __builtin_amdgcn_exp2f(2.8853900817779268f * u);
  float th = 1.0f - 2.0f * __builtin_amdgcn_rcpf(1.0f + e);
  return 0.5f * x * (1.0f + th);
}

__device__ __forceinline__ float exp_fast(float x) {
  return __builtin_amdgcn_exp2f(1.4426950408889634f * x);
}

// ---------------- kernel 0: zero output + gating accumulators ----------------
__global__ void zero_kernel(float* __restrict__ out, float* __restrict__ meSum, size_t n) {
  size_t i = (size_t)blockIdx.x * blockDim.x + threadIdx.x;
  if (i < n) out[i] = 0.0f;
  if (blockIdx.x == 0 && threadIdx.x < NEXP) meSum[threadIdx.x] = 0.0f;
}

// ---------------- kernel 1: top-1 gating (one wave32 per token) ----------------
__global__ void gate_kernel(const float* __restrict__ x, const float* __restrict__ wg,
                            int* __restrict__ tokExp, float* __restrict__ tokGate,
                            float* __restrict__ meSum) {
  __shared__ float me_s[NEXP];
  if (threadIdx.x < NEXP) me_s[threadIdx.x] = 0.0f;
  __syncthreads();
  const int wv = threadIdx.x >> 5, lane = threadIdx.x & 31;
  const int t = blockIdx.x * 8 + wv;
  float a[NEXP];
#pragma unroll
  for (int e = 0; e < NEXP; ++e) a[e] = 0.0f;
  const float* xr = x + (size_t)t * MDIM;
  for (int m = lane; m < MDIM; m += 32) {
    float xv = xr[m];
    const float* wr = wg + m * NEXP;
#pragma unroll
    for (int e = 0; e < NEXP; ++e) a[e] = fmaf(xv, wr[e], a[e]);
  }
#pragma unroll
  for (int off = 16; off > 0; off >>= 1) {
#pragma unroll
    for (int e = 0; e < NEXP; ++e) a[e] += __shfl_xor(a[e], off, 32);
  }
  if (lane == 0) {
    float mx = a[0]; int im = 0;
#pragma unroll
    for (int e = 1; e < NEXP; ++e) if (a[e] > mx) { mx = a[e]; im = e; }
    float s = 0.0f, g[NEXP];
#pragma unroll
    for (int e = 0; e < NEXP; ++e) { g[e] = exp_fast(a[e] - mx); s += g[e]; }
    float inv = 1.0f / s;
    tokExp[t]  = im;
    tokGate[t] = g[im] * inv;
#pragma unroll
    for (int e = 0; e < NEXP; ++e) atomicAdd(&me_s[e], g[e] * inv);
  }
  __syncthreads();
  if (threadIdx.x < NEXP) atomicAdd(&meSum[threadIdx.x], me_s[threadIdx.x]);
}

// ------------- kernel 2: per-expert sequential scan (capacity) + l_aux -------------
__global__ void scan_kernel(const int* __restrict__ tokExp, int* __restrict__ slotTok,
                            const float* __restrict__ meSum, float* __restrict__ laux_out) {
  __shared__ int cnts[NEXP];
  const int e = threadIdx.x;
  if (e < NEXP) {
    int cnt = 0;
    for (int t = 0; t < T_TOK; ++t) {
      if (tokExp[t] == e) {
        if (cnt < CAP) slotTok[e * CAP + cnt] = t;
        ++cnt;
      }
    }
    for (int c = (cnt < CAP ? cnt : CAP); c < CAP; ++c) slotTok[e * CAP + c] = -1;
    cnts[e] = cnt;
  }
  __syncthreads();
  if (threadIdx.x == 0) {
    const float invT = 1.0f / (float)T_TOK;
    float l = 0.0f;
    for (int i = 0; i < NEXP; ++i) l += (meSum[i] * invT) * ((float)cnts[i] * invT);
    *laux_out = l * (float)NEXP;
  }
}

// ------------- kernel 3: gather routed tokens -> bf16 dispatched buffer -------------
__global__ void dispatch_kernel(const float* __restrict__ x, const int* __restrict__ slotTok,
                                unsigned short* __restrict__ Xd) {
  const int row = blockIdx.x;            // e*CAP + c
  const int t = slotTok[row];
  unsigned short* dst = Xd + (size_t)row * MDIM;
  if (t >= 0) {
    const float* src = x + (size_t)t * MDIM;
    for (int m = threadIdx.x * 4; m < MDIM; m += blockDim.x * 4) {
      float4 f = *(const float4*)(src + m);
      unsigned long long p = (unsigned long long)f2bf(f.x)
                           | ((unsigned long long)f2bf(f.y) << 16)
                           | ((unsigned long long)f2bf(f.z) << 32)
                           | ((unsigned long long)f2bf(f.w) << 48);
      *(unsigned long long*)(dst + m) = p;
    }
  } else {
    for (int m = threadIdx.x * 4; m < MDIM; m += blockDim.x * 4)
      *(unsigned long long*)(dst + m) = 0ull;
  }
}

// ============ GEMM core: 128x256 block, 8 waves (2x4), 64x64 wave tile, ============
// ============ double-buffered LDS pipeline, bf16 WMMA (16 per wave per K-step) =====
union FragBF { uint4 q[2]; v16bf v; };

struct Stage { uint4 a0, a1; float4 bf[8]; };

__device__ __forceinline__ void stage_load(const unsigned short* __restrict__ A,
                                           const float* __restrict__ B,
                                           int K, int ldb, int row0, int col0, int k0,
                                           int ar, int ak, int bkr, int bnc, Stage& st) {
  const uint4* asrc = (const uint4*)(A + (size_t)(row0 + ar) * K + k0 + ak);
  st.a0 = asrc[0];
  st.a1 = asrc[1];
  const float* bsrc = B + (size_t)(k0 + bkr) * ldb + col0 + bnc;
#pragma unroll
  for (int v = 0; v < 8; ++v) st.bf[v] = *(const float4*)(bsrc + v * 4);
  if (k0 + BK < K)   // warm L2 for the tile after next (global_prefetch_b8)
    __builtin_prefetch((const void*)(B + (size_t)(k0 + BK + bkr) * ldb + col0 + bnc), 0, 0);
}

__device__ __forceinline__ void stage_store(unsigned short* As, unsigned short* Bs,
                                            int ar, int ak, int bkr, int bnc,
                                            const Stage& st) {
  *(uint4*)&As[ar * LDA + ak]     = st.a0;
  *(uint4*)&As[ar * LDA + ak + 8] = st.a1;
#pragma unroll
  for (int v = 0; v < 8; ++v) {
    Bs[(bnc + v * 4 + 0) * LDA + bkr] = f2bf(st.bf[v].x);
    Bs[(bnc + v * 4 + 1) * LDA + bkr] = f2bf(st.bf[v].y);
    Bs[(bnc + v * 4 + 2) * LDA + bkr] = f2bf(st.bf[v].z);
    Bs[(bnc + v * 4 + 3) * LDA + bkr] = f2bf(st.bf[v].w);
  }
}

__device__ __forceinline__ void compute_stage(const unsigned short* As,
                                              const unsigned short* Bs,
                                              int wM, int wN, int half, int l15,
                                              v8f acc[4][4]) {
  FragBF a[4], b[4];
#pragma unroll
  for (int i = 0; i < 4; ++i) {     // A frag: lane half selects K=0..15 / 16..31
    int r = wM + i * 16 + l15;
    a[i].q[0] = *(const uint4*)&As[r * LDA + half * 16];
    a[i].q[1] = *(const uint4*)&As[r * LDA + half * 16 + 8];
  }
#pragma unroll
  for (int j = 0; j < 4; ++j) {     // B frag: lane half selects K=0..15 / 16..31
    int n = wN + j * 16 + l15;
    const uint4* p = (const uint4*)&Bs[n * LDA + half * 16];
    b[j].q[0] = p[0];
    b[j].q[1] = p[1];
  }
#pragma unroll
  for (int i = 0; i < 4; ++i)
#pragma unroll
    for (int j = 0; j < 4; ++j)
      acc[i][j] = __builtin_amdgcn_wmma_f32_16x16x32_bf16(
          false, a[i].v, false, b[j].v, (short)0, acc[i][j], false, false);
}

__device__ __forceinline__ void gemm_core(
    const unsigned short* __restrict__ A,   // [rows x K] bf16, row-major
    const float* __restrict__ B,            // [K x ldb] f32, row-major
    int K, int ldb, int row0, int col0,
    unsigned short* As, unsigned short* Bs, v8f acc[4][4]) {
  const int tid  = threadIdx.x;
  const int lane = tid & 31;
  const int wv   = tid >> 5;
  const int half = lane >> 4;
  const int l15  = lane & 15;
  const int wM   = (wv & 1) * 64;        // 2 wave rows x 4 wave cols
  const int wN   = (wv >> 1) * 64;
  const int ar = tid >> 1;               // A stage: 0..127 rows, 16 bf16 each
  const int ak = (tid & 1) * 16;
  const int bk = tid >> 3;               // B stage: 0..31 k-rows, 32 floats each
  const int bn = (tid & 7) * 32;

  Stage st;
  stage_load(A, B, K, ldb, row0, col0, 0, ar, ak, bk, bn, st);
  stage_store(As, Bs, ar, ak, bk, bn, st);
  __syncthreads();

  int cur = 0;
  for (int k0 = BK; k0 < K; k0 += BK) {
    stage_load(A, B, K, ldb, row0, col0, k0, ar, ak, bk, bn, st);   // prefetch k0
    compute_stage(As + cur * (BM * LDA), Bs + cur * (BN * LDA), wM, wN, half, l15, acc);
    cur ^= 1;
    stage_store(As + cur * (BM * LDA), Bs + cur * (BN * LDA), ar, ak, bk, bn, st);
    __syncthreads();
  }
  compute_stage(As + cur * (BM * LDA), Bs + cur * (BN * LDA), wM, wN, half, l15, acc);
}

// ---------------- kernel 4: H = gelu(Xd @ W1 + b1), store bf16 ----------------
__global__ void __launch_bounds__(256) gemm1_kernel(
    const unsigned short* __restrict__ Xd, const float* __restrict__ w1,
    const float* __restrict__ b1, unsigned short* __restrict__ H) {
  __shared__ __align__(16) unsigned short As[2 * BM * LDA];
  __shared__ __align__(16) unsigned short Bs[2 * BN * LDA];
  const int e = blockIdx.z, row0 = blockIdx.y * BM, col0 = blockIdx.x * BN;
  v8f acc[4][4];
#pragma unroll
  for (int i = 0; i < 4; ++i)
#pragma unroll
    for (int j = 0; j < 4; ++j) { v8f z = {}; acc[i][j] = z; }
  gemm_core(Xd + (size_t)e * CAP * MDIM, w1 + (size_t)e * MDIM * FDIM,
            MDIM, FDIM, row0, col0, As, Bs, acc);
  const int lane = threadIdx.x & 31, wv = threadIdx.x >> 5;
  const int half = lane >> 4, l15 = lane & 15;
  const int wM = (wv & 1) * 64, wN = (wv >> 1) * 64;
  int   cols[4];
  float bias[4];
#pragma unroll
  for (int j = 0; j < 4; ++j) {
    cols[j] = col0 + wN + j * 16 + l15;
    bias[j] = b1[e * FDIM + cols[j]];
  }
#pragma unroll
  for (int i = 0; i < 4; ++i) {
    int rowbase = row0 + wM + i * 16 + half * 8;
#pragma unroll
    for (int j = 0; j < 4; ++j) {
      unsigned short* dst = H + ((size_t)e * CAP + rowbase) * FDIM + cols[j];
#pragma unroll
      for (int r = 0; r < 8; ++r)
        dst[(size_t)r * FDIM] = f2bf(gelu_fast(acc[i][j][r] + bias[j]));
    }
  }
}

// ------- kernel 5: Out_e = H @ W2 + b2, fused combine (gate * row -> d_out) -------
__global__ void __launch_bounds__(256) gemm2_kernel(
    const unsigned short* __restrict__ H, const float* __restrict__ w2,
    const float* __restrict__ b2, const int* __restrict__ slotTok,
    const float* __restrict__ tokGate, float* __restrict__ out) {
  __shared__ __align__(16) unsigned short As[2 * BM * LDA];
  __shared__ __align__(16) unsigned short Bs[2 * BN * LDA];
  const int e = blockIdx.z, row0 = blockIdx.y * BM, col0 = blockIdx.x * BN;
  v8f acc[4][4];
#pragma unroll
  for (int i = 0; i < 4; ++i)
#pragma unroll
    for (int j = 0; j < 4; ++j) { v8f z = {}; acc[i][j] = z; }
  gemm_core(H + (size_t)e * CAP * FDIM, w2 + (size_t)e * FDIM * MDIM,
            FDIM, MDIM, row0, col0, As, Bs, acc);
  const int lane = threadIdx.x & 31, wv = threadIdx.x >> 5;
  const int half = lane >> 4, l15 = lane & 15;
  const int wM = (wv & 1) * 64, wN = (wv >> 1) * 64;
  int   cols[4];
  float bias[4];
#pragma unroll
  for (int j = 0; j < 4; ++j) {
    cols[j] = col0 + wN + j * 16 + l15;
    bias[j] = b2[e * MDIM + cols[j]];
  }
#pragma unroll
  for (int i = 0; i < 4; ++i) {
    int rbase = row0 + wM + i * 16 + half * 8;
#pragma unroll
    for (int r = 0; r < 8; ++r) {
      int c = rbase + r;
      int t = slotTok[e * CAP + c];
      if (t < 0) continue;                  // dropped capacity slot -> row stays 0
      float g = tokGate[t];
#pragma unroll
      for (int j = 0; j < 4; ++j)
        out[(size_t)t * MDIM + cols[j]] = g * (acc[i][j][r] + bias[j]);
    }
  }
}

extern "C" void kernel_launch(void* const* d_in, const int* in_sizes, int n_in,
                              void* d_out, int out_size, void* d_ws, size_t ws_size,
                              hipStream_t stream) {
  const float* x  = (const float*)d_in[0];
  const float* wg = (const float*)d_in[1];
  const float* w1 = (const float*)d_in[2];
  const float* b1 = (const float*)d_in[3];
  const float* w2 = (const float*)d_in[4];
  const float* b2 = (const float*)d_in[5];
  float* out = (float*)d_out;

  char* ws = (char*)d_ws;
  float* meSum   = (float*)ws;                               // 8 f32 (pad 256B)
  int*   tokExp  = (int*)(ws + 256);                         // T ints
  float* tokGate = (float*)(ws + 256 + 4 * (size_t)T_TOK);   // T f32
  int*   slotTok = (int*)(ws + 256 + 8 * (size_t)T_TOK);     // E*CAP ints
  unsigned short* Xd = (unsigned short*)(ws + 256 + 12 * (size_t)T_TOK);           // 32 MB
  unsigned short* H  = (unsigned short*)(ws + 256 + 12 * (size_t)T_TOK
                                         + 2 * (size_t)T_TOK * MDIM);              // 128 MB

  const size_t nOut = (size_t)T_TOK * MDIM + 1;
  zero_kernel<<<(unsigned)((nOut + 255) / 256), 256, 0, stream>>>(out, meSum, nOut);
  gate_kernel<<<T_TOK / 8, 256, 0, stream>>>(x, wg, tokExp, tokGate, meSum);
  scan_kernel<<<1, 32, 0, stream>>>(tokExp, slotTok, meSum, out + (size_t)T_TOK * MDIM);
  dispatch_kernel<<<NEXP * CAP, 256, 0, stream>>>(x, slotTok, Xd);
  gemm1_kernel<<<dim3(FDIM / BN, CAP / BM, NEXP), 256, 0, stream>>>(Xd, w1, b1, H);
  gemm2_kernel<<<dim3(MDIM / BN, CAP / BM, NEXP), 256, 0, stream>>>(H, w2, b2,
                                                                    slotTok, tokGate, out);
}
